// SparseDecoder_79559974191776
// MI455X (gfx1250) — matrix operand order repaired and checked by reading, not compiled, not run on hardware
//
#include <hip/hip_runtime.h>
#include <math.h>

typedef _Float16 half8 __attribute__((ext_vector_type(8)));
typedef _Float16 v16h  __attribute__((ext_vector_type(16)));
typedef float    v8f   __attribute__((ext_vector_type(8)));

static constexpr int BATCH = 2048;
static constexpr int DZ    = 64;
static constexpr int INT_C = 256;
static constexpr float EPS = 1e-5f;
static constexpr int LDS_PAD = 8;   // halfs; 16B pad -> conflict-free row-strided ds reads

// ---------------------------------------------------------------- helpers

__device__ __forceinline__ v16h mkv16(half8 lo, half8 hi) {
  v16h r;
#pragma unroll
  for (int i = 0; i < 8; ++i) { r[i] = lo[i]; r[i + 8] = hi[i]; }
  return r;
}

// A fragment (16x32 f16): lane l (m = l&15, g = l>>4) holds
//   a[j] = A[m, kbase + g*16 + j], j = 0..15
__device__ __forceinline__ v16h load_a_frag(const _Float16* rowp, int koff) {
  half8 lo = *(const half8*)(rowp + koff);
  half8 hi = *(const half8*)(rowp + koff + 8);
  return mkv16(lo, hi);
}

// B fragment (32x16 f16): lane l (n = l&15, g = l>>4) holds b[j] = B[kbase+g*16+j, n]
// colp = &Wt[n*Kdim + kbase + g*16], Wt = B^T [N][K]
__device__ __forceinline__ v16h load_b_frag(const _Float16* colp) {
  half8 lo = *(const half8*)(colp);
  half8 hi = *(const half8*)(colp + 8);
  return mkv16(lo, hi);
}

__device__ __forceinline__ v8f wmma16(v16h a, v16h b, v8f c) {
  return __builtin_amdgcn_wmma_f32_16x16x32_f16(false, a, false, b,
                                                (short)0, c, false, false);
}

__device__ __forceinline__ float lrelu(float v) { return v > 0.0f ? v : 0.2f * v; }

// ---------------------------------------------------------------- prep kernels

__global__ void k_cvt_f16(const float* __restrict__ s, _Float16* __restrict__ d, int n) {
  int i = blockIdx.x * 256 + threadIdx.x;
  if (i < n) d[i] = (_Float16)s[i];
}

// dense GEMM B^T: dst[n*K + k] = src[k*N + n]  (f32 -> f16)
__global__ void k_transpose_f16(const float* __restrict__ s, _Float16* __restrict__ d,
                                int K, int N) {
  int i = blockIdx.x * 256 + threadIdx.x;
  if (i >= K * N) return;
  int n = i / K, k = i % K;
  d[i] = (_Float16)s[(size_t)k * N + n];
}

// convT weights: src [Cin][Cout][KK] -> dst [KK][Cout][Cin]  (per-tap B^T)
__global__ void k_pack_wT(const float* __restrict__ s, _Float16* __restrict__ d,
                          int Cin, int Cout, int KK) {
  int i = blockIdx.x * 256 + threadIdx.x;
  if (i >= Cin * Cout * KK) return;
  int ci = i % Cin;
  int o  = (i / Cin) % Cout;
  int kk = i / (Cin * Cout);
  d[i] = (_Float16)s[((size_t)ci * Cout + o) * KK + kk];
}

// ---------------------------------------------------------------- L1: [2048,64]x[64,256]
// 8 waves x 2 N-subtiles = full N=256 per block row; A fragment reused for 2 WMMAs.

__global__ __launch_bounds__(256) void k_gemm_l1(
    const _Float16* __restrict__ Z, const _Float16* __restrict__ W1t,
    const float* __restrict__ b1, const float* __restrict__ g1,
    const float* __restrict__ be1, _Float16* __restrict__ X1) {
  const int lane = threadIdx.x & 31, wid = threadIdx.x >> 5;
  const int m = lane & 15, g = lane >> 4;
  const int n0 = (wid * 2) * 16 + m, n1 = n0 + 16;
  const int arow = blockIdx.x * 16 + m;
  const _Float16* ap  = Z + (size_t)arow * DZ;
  const _Float16* bp0 = W1t + (size_t)n0 * DZ + g * 16;
  const _Float16* bp1 = W1t + (size_t)n1 * DZ + g * 16;
  v8f acc0 = {}, acc1 = {};
#pragma unroll
  for (int kb = 0; kb < DZ; kb += 32) {
    v16h a = load_a_frag(ap, kb + g * 16);
    acc0 = wmma16(a, load_b_frag(bp0 + kb), acc0);
    acc1 = wmma16(a, load_b_frag(bp1 + kb), acc1);
  }
#pragma unroll
  for (int s = 0; s < 2; ++s) {
    const int n = s ? n1 : n0;
    const v8f acc = s ? acc1 : acc0;
    const float sc = g1[n] * rsqrtf(1.0f + EPS);
    const float bi = b1[n], bt = be1[n];
#pragma unroll
    for (int r = 0; r < 8; ++r) {
      int rr = blockIdx.x * 16 + g * 8 + r;
      float v = lrelu((acc[r] + bi) * sc + bt);
      X1[(size_t)rr * INT_C + n] = (_Float16)v;
    }
  }
}

// ------------------------------------ L2: [2048,256]x[256,12544] -> NHWC [2048,7,7,256]
// 8 waves x 2 subtiles = 256 columns per block; grid.y covers 12544/256 = 49.

__global__ __launch_bounds__(256) void k_gemm_l2(
    const _Float16* __restrict__ X1, const _Float16* __restrict__ W2t,
    const float* __restrict__ b2, const float* __restrict__ g2,
    const float* __restrict__ be2, _Float16* __restrict__ X2) {
  const int lane = threadIdx.x & 31, wid = threadIdx.x >> 5;
  const int m = lane & 15, g = lane >> 4;
  const int n0 = (blockIdx.y * 16 + wid * 2) * 16 + m, n1 = n0 + 16;  // n = o*49+p
  const int arow = blockIdx.x * 16 + m;
  const _Float16* ap  = X1 + (size_t)arow * INT_C;
  const _Float16* bp0 = W2t + (size_t)n0 * INT_C + g * 16;
  const _Float16* bp1 = W2t + (size_t)n1 * INT_C + g * 16;
  v8f acc0 = {}, acc1 = {};
#pragma unroll
  for (int kb = 0; kb < INT_C; kb += 32) {
    v16h a = load_a_frag(ap, kb + g * 16);
    acc0 = wmma16(a, load_b_frag(bp0 + kb), acc0);
    acc1 = wmma16(a, load_b_frag(bp1 + kb), acc1);
  }
#pragma unroll
  for (int s = 0; s < 2; ++s) {
    const int n = s ? n1 : n0;
    const v8f acc = s ? acc1 : acc0;
    const int o = n / 49, p = n % 49;
    const float sc = g2[o] * rsqrtf(1.0f + EPS);
    const float bi = b2[o], bt = be2[o];
#pragma unroll
    for (int r = 0; r < 8; ++r) {
      int rr = blockIdx.x * 16 + g * 8 + r;        // batch index
      float v = lrelu((acc[r] + bi) * sc + bt);
      X2[((size_t)rr * 49 + p) * INT_C + o] = (_Float16)v;
    }
  }
}

// ---------------- L3/L4: 4x4 stride-2 pad-1 convT, parity-class implicit gather-GEMM.
// grid.y = parity class (py*2+px): all 4 taps (ky,kx) are block-uniform.
// The gathered 16-row A tile (4 taps x 16 rows x Cin, invalid rows zero-filled) is staged
// cooperatively into LDS once, then all waves read fragments via ds_load.
// blockDim = Cout threads = Cout/32 waves, each wave covers 2 N-subtiles.

__global__ void k_convt_wmma(
    const _Float16* __restrict__ Xin, const _Float16* __restrict__ WpT,
    const float* __restrict__ bias, const float* __restrict__ gam,
    const float* __restrict__ bet, _Float16* __restrict__ Yout,
    int Hin, int Win, int Cin, int Cout) {
  extern __shared__ _Float16 smem[];               // [4][16][Cin+LDS_PAD]
  const int ldsr = Cin + LDS_PAD;
  const int lane = threadIdx.x & 31, wid = threadIdx.x >> 5;
  const int m = lane & 15, g = lane >> 4;
  const int py = blockIdx.y >> 1, px = blockIdx.y & 1;
  const int HW = Hin * Win;

  // ---- cooperative gather: global (NHWC rows, tap-dependent) -> LDS, 16B chunks
  {
    const int cpr = Cin / 8;                       // h8 chunks per row
    const int cpt = 16 * cpr;                      // chunks per tap
    const int total = 4 * cpt;
    for (int c = threadIdx.x; c < total; c += blockDim.x) {
      const int t  = c / cpt;
      const int r  = (c % cpt) / cpr;
      const int k8 = c % cpr;
      const int dy = t >> 1, dx = t & 1;
      const int row = blockIdx.x * 16 + r;
      const int ab = row / HW, apix = row % HW;
      const int iy = apix / Win + py - dy;
      const int ix = apix % Win + px - dx;
      half8 v = {};
      if ((unsigned)iy < (unsigned)Hin && (unsigned)ix < (unsigned)Win)
        v = *(const half8*)(Xin + (((size_t)ab * Hin + iy) * Win + ix) * Cin + k8 * 8);
      *(half8*)(smem + ((size_t)t * 16 + r) * ldsr + k8 * 8) = v;
    }
  }
  __syncthreads();

  // ---- WMMA main loop: A from LDS (shared), 2 B columns per wave
  const int n0 = (wid * 2) * 16 + m, n1 = n0 + 16;
  v8f acc0 = {}, acc1 = {};
#pragma unroll
  for (int t = 0; t < 4; ++t) {
    const int dy = t >> 1, dx = t & 1;
    const int ky = (1 - py) + 2 * dy;              // ky ≡ (oy+1) mod 2
    const int kx = (1 - px) + 2 * dx;
    const _Float16* arow = smem + ((size_t)t * 16 + m) * ldsr;
    const _Float16* bp0 = WpT + ((size_t)(ky * 4 + kx) * Cout + n0) * Cin + g * 16;
    const _Float16* bp1 = WpT + ((size_t)(ky * 4 + kx) * Cout + n1) * Cin + g * 16;
    for (int kb = 0; kb < Cin; kb += 32) {
      v16h a = load_a_frag(arow, kb + g * 16);     // ds_load_b128 x2
      acc0 = wmma16(a, load_b_frag(bp0 + kb), acc0);
      acc1 = wmma16(a, load_b_frag(bp1 + kb), acc1);
    }
  }

  // ---- epilogue: bias + BN(eval) + LeakyReLU -> NHWC f16
  const int Hout = 2 * Hin, Wout = 2 * Win;
#pragma unroll
  for (int s = 0; s < 2; ++s) {
    const int n = s ? n1 : n0;
    const v8f acc = s ? acc1 : acc0;
    const float sc = gam[n] * rsqrtf(1.0f + EPS);
    const float bi = bias[n], bt = bet[n];
#pragma unroll
    for (int r = 0; r < 8; ++r) {
      int rr = blockIdx.x * 16 + g * 8 + r;
      int bb = rr / HW, pp = rr % HW;
      int oy = 2 * (pp / Win) + py, ox = 2 * (pp % Win) + px;
      float v = lrelu((acc[r] + bi) * sc + bt);
      Yout[(((size_t)bb * Hout + oy) * Wout + ox) * Cout + n] = (_Float16)v;
    }
  }
}

// ---------------- L5: 3x3 stride-1 pad-1 convT, Cout=1, + sigmoid (VALU; N=1 not WMMA-shaped)

__global__ void k_conv5_sigmoid(const _Float16* __restrict__ X4,
                                const _Float16* __restrict__ W5p,  // [9][64]
                                const float* __restrict__ b5, float* __restrict__ out) {
  int idx = blockIdx.x * 256 + threadIdx.x;
  if (idx >= BATCH * 28 * 28) return;
  int b = idx / 784, p = idx % 784, oy = p / 28, ox = p % 28;
  float acc = b5[0];
  for (int ky = 0; ky < 3; ++ky) {
    int iy = oy + 1 - ky;
    if ((unsigned)iy >= 28u) continue;
    for (int kx = 0; kx < 3; ++kx) {
      int ix = ox + 1 - kx;
      if ((unsigned)ix >= 28u) continue;
      const _Float16* xp = X4 + (((size_t)b * 28 + iy) * 28 + ix) * 64;
      const _Float16* wp = W5p + (ky * 3 + kx) * 64;
#pragma unroll
      for (int c = 0; c < 64; ++c) acc += (float)xp[c] * (float)wp[c];
    }
  }
  out[idx] = 1.0f / (1.0f + expf(-acc));
}

// ---------------------------------------------------------------- launcher

extern "C" void kernel_launch(void* const* d_in, const int* in_sizes, int n_in,
                              void* d_out, int out_size, void* d_ws, size_t ws_size,
                              hipStream_t stream) {
  (void)in_sizes; (void)n_in; (void)out_size; (void)ws_size;
  const float* z   = (const float*)d_in[0];
  const float* w1  = (const float*)d_in[1];
  const float* b1  = (const float*)d_in[2];
  const float* g1  = (const float*)d_in[3];
  const float* be1 = (const float*)d_in[4];
  const float* w2  = (const float*)d_in[5];
  const float* b2  = (const float*)d_in[6];
  const float* g2  = (const float*)d_in[7];
  const float* be2 = (const float*)d_in[8];
  const float* w3  = (const float*)d_in[9];
  const float* b3  = (const float*)d_in[10];
  const float* g3  = (const float*)d_in[11];
  const float* be3 = (const float*)d_in[12];
  const float* w4  = (const float*)d_in[13];
  const float* b4  = (const float*)d_in[14];
  const float* g4  = (const float*)d_in[15];
  const float* be4 = (const float*)d_in[16];
  const float* w5  = (const float*)d_in[17];
  const float* b5  = (const float*)d_in[18];

  char* ws = (char*)d_ws;
  size_t off = 0;
  auto alloc = [&](size_t halfs) {
    _Float16* p = (_Float16*)(ws + off);
    off += ((halfs * 2 + 255) & ~(size_t)255);
    return p;
  };
  _Float16* zh  = alloc((size_t)BATCH * DZ);
  _Float16* w1t = alloc((size_t)DZ * INT_C);
  _Float16* w2t = alloc((size_t)INT_C * 12544);
  _Float16* w3t = alloc((size_t)16 * 256 * 128);
  _Float16* w4t = alloc((size_t)16 * 128 * 64);
  _Float16* w5p = alloc((size_t)9 * 64);
  _Float16* x1  = alloc((size_t)BATCH * INT_C);
  _Float16* x2  = alloc((size_t)BATCH * 49 * 256);
  _Float16* x3  = alloc((size_t)BATCH * 196 * 128);
  _Float16* x4  = alloc((size_t)BATCH * 784 * 64);

  // ---- weight / input preparation (f32 -> f16, B^T layouts)
  k_cvt_f16<<<(BATCH * DZ + 255) / 256, 256, 0, stream>>>(z, zh, BATCH * DZ);
  k_transpose_f16<<<(DZ * INT_C + 255) / 256, 256, 0, stream>>>(w1, w1t, DZ, INT_C);
  k_transpose_f16<<<(INT_C * 12544 + 255) / 256, 256, 0, stream>>>(w2, w2t, INT_C, 12544);
  k_pack_wT<<<(16 * 256 * 128 + 255) / 256, 256, 0, stream>>>(w3, w3t, 256, 128, 16);
  k_pack_wT<<<(16 * 128 * 64 + 255) / 256, 256, 0, stream>>>(w4, w4t, 128, 64, 16);
  k_pack_wT<<<(9 * 64 + 255) / 256, 256, 0, stream>>>(w5, w5p, 64, 1, 9);

  // ---- decode pipeline (stream-ordered)
  k_gemm_l1<<<dim3(BATCH / 16, 1), 256, 0, stream>>>(zh, w1t, b1, g1, be1, x1);
  k_gemm_l2<<<dim3(BATCH / 16, 12544 / 256), 256, 0, stream>>>(x1, w2t, b2, g2, be2, x2);
  // L3: M_class = 2048*49, Cout=128 -> 128 threads (4 waves x 2 subtiles), 33 KB LDS
  k_convt_wmma<<<dim3((BATCH * 49) / 16, 4), 128,
                 4 * 16 * (256 + LDS_PAD) * sizeof(_Float16), stream>>>(
      x2, w3t, b3, g3, be3, x3, 7, 7, 256, 128);
  // L4: M_class = 2048*196, Cout=64 -> 64 threads (2 waves x 2 subtiles), 17 KB LDS
  k_convt_wmma<<<dim3((BATCH * 196) / 16, 4), 64,
                 4 * 16 * (128 + LDS_PAD) * sizeof(_Float16), stream>>>(
      x3, w4t, b4, g4, be4, x4, 14, 14, 128, 64);
  k_conv5_sigmoid<<<(BATCH * 784 + 255) / 256, 256, 0, stream>>>(x4, w5p, b5, (float*)d_out);
}
